// HFR_66889820668188
// MI455X (gfx1250) — compile-verified, matrix operand order checked by reading, not run on hardware
//
#include <hip/hip_runtime.h>
#include <hip/hip_bf16.h>

// ---------- types ----------
typedef __attribute__((ext_vector_type(16))) __bf16 v16bf;
typedef __attribute__((ext_vector_type(8)))  float  v8f;
typedef __attribute__((ext_vector_type(4)))  int    vi4;

union V16U { v16bf v; vi4 q[2]; };

#define IN_F   1024
#define OUT_F  1024
#define GRID_G 7
#define KTOT   8192          // 1024 (silu part) + 7168 (wavelet part)
#define NROWS  16384         // 4 * 4096

// ---------------------------------------------------------------------------
// Kernel 0: spectral norm.  sigma = ||W v|| (with eps), v = normalize(W^T u).
// One 1024-thread block; writes 1/sigma to workspace.
// ---------------------------------------------------------------------------
__global__ __launch_bounds__(1024)
void k_sigma(const float* __restrict__ W, const float* __restrict__ u,
             float* __restrict__ inv_sigma) {
  __shared__ float red[1024];
  __shared__ float sv[1024];
  const int t = threadIdx.x;

  // w = W^T u  (thread t owns column t; reads coalesced across threads)
  float w = 0.f;
  for (int o = 0; o < OUT_F; ++o) w += W[(size_t)o * IN_F + t] * u[o];
  red[t] = w * w;
  __syncthreads();
  for (int s = 512; s > 0; s >>= 1) { if (t < s) red[t] += red[t + s]; __syncthreads(); }
  float nv = sqrtf(red[0]);
  sv[t] = w / (nv + 1e-12f);
  __syncthreads();

  // y = W v  (thread t owns row t)
  float y = 0.f;
  for (int i = 0; i < IN_F; ++i) y += W[(size_t)t * IN_F + i] * sv[i];
  red[t] = y * y;
  __syncthreads();
  for (int s = 512; s > 0; s >>= 1) { if (t < s) red[t] += red[t + s]; __syncthreads(); }
  if (t == 0) {
    float ns    = sqrtf(red[0]);
    float sigma = ns * ns / (ns + 1e-12f);
    *inv_sigma  = 1.0f / sigma;
  }
}

// ---------------------------------------------------------------------------
// Kernel 1: pack concatenated bf16 weights: Wp[o][k], k<1024 -> base_w/sigma,
// else wavelet_w (reshape col j = i*7+g is already the memory order).
// ---------------------------------------------------------------------------
__global__ __launch_bounds__(256)
void k_pack(const float* __restrict__ base_w, const float* __restrict__ wavelet_w,
            const float* __restrict__ inv_sigma, __bf16* __restrict__ Wp) {
  size_t idx = (size_t)blockIdx.x * 256 + threadIdx.x;
  if (idx >= (size_t)OUT_F * KTOT) return;
  int o = (int)(idx / KTOT);
  int k = (int)(idx % KTOT);
  float v;
  if (k < IN_F) v = base_w[(size_t)o * IN_F + k] * (*inv_sigma);
  else          v = wavelet_w[(size_t)o * (IN_F * GRID_G) + (k - IN_F)];
  Wp[idx] = (__bf16)v;
}

// ---------------------------------------------------------------------------
// Kernel 2: fused dual-GEMM via v_wmma_f32_16x16x32_bf16.
// Block = 256 threads (8 waves) -> 16 rows x 128 outputs; wave -> 16x16 tile.
// ---------------------------------------------------------------------------
__global__ __launch_bounds__(256)
void k_main(const float* __restrict__ x, const __bf16* __restrict__ Wp,
            const float* __restrict__ bias, const float* __restrict__ translation,
            const float* __restrict__ scale, const float* __restrict__ soft_threshold,
            const float* __restrict__ output_scale, float* __restrict__ out) {
  __shared__ __align__(16) __bf16 sA[16][232];   // 224 used, 232 stride (16B-mult)

  const int t     = threadIdx.x;
  const int lane  = t & 31;
  const int wave  = t >> 5;
  const int mbase = blockIdx.x * 16;
  const int obase = blockIdx.y * 128 + wave * 16;
  const int n     = lane & 15;      // output column within tile / A row
  const int hi    = lane >> 4;      // K-half select per WMMA layout

  // Chebyshev translation points + fixed scales (7 each) -> registers
  float tr[GRID_G], is[GRID_G];
#pragma unroll
  for (int g = 0; g < GRID_G; ++g) {
    tr[g] = translation[g];
    float s = fmaxf(fabsf(scale[g]), 0.1f);
    is[g] = 1.0f / s;
  }

  v8f accB = {};   // silu(x) @ w_sn^T
  v8f accK = {};   // basis  @ wavelet_w^T

  const __bf16* wrow = Wp + (size_t)(obase + n) * KTOT;

  // ---- Part 1: K = 1024 (silu), chunks of 128 ----
  for (int kc = 0; kc < IN_F; kc += 128) {
    __syncthreads();
#pragma unroll
    for (int e = 0; e < 8; ++e) {               // 16x128 elems, 8 per thread
      int el = e * 256 + t;
      int m  = el >> 7;
      int k  = el & 127;
      float v = x[(size_t)(mbase + m) * IN_F + kc + k];
      sA[m][k] = (__bf16)(v / (1.0f + __expf(-v)));   // silu
    }
    __syncthreads();
#pragma unroll
    for (int kk = 0; kk < 128; kk += 32) {
      V16U a, b;
      const __bf16* ap = &sA[n][kk + hi * 8];
      a.q[0] = *(const vi4*)(ap);
      a.q[1] = *(const vi4*)(ap + 16);
      const __bf16* bp = wrow + kc + kk + hi * 8;
      b.q[0] = *(const vi4*)(bp);
      b.q[1] = *(const vi4*)(bp + 16);
      accB = __builtin_amdgcn_wmma_f32_16x16x32_bf16(false, a.v, false, b.v,
                                                     (short)0, accB, false, false);
    }
  }

  // ---- Part 2: K = 7168 (Ricker basis), chunks of 224 = 32 feats * 7 ----
  for (int c = 0; c < 32; ++c) {
    __syncthreads();
#pragma unroll
    for (int p = 0; p < 2; ++p) {               // 512 (m,feature) pairs, 2/thread
      int pair = p * 256 + t;
      int m = pair & 15;
      int f = pair >> 4;                        // 0..31
      float v  = x[(size_t)(mbase + m) * IN_F + c * 32 + f];
      float tn = tanhf(v) * 2.5f;
#pragma unroll
      for (int g = 0; g < GRID_G; ++g) {
        float xe = (tn - tr[g]) * is[g];
        float x2 = xe * xe;
        sA[m][f * 7 + g] = (__bf16)((1.0f - x2) * __expf(-0.5f * x2));
      }
    }
    __syncthreads();
    const __bf16* wrow2 = wrow + IN_F + c * 224;
#pragma unroll
    for (int kk = 0; kk < 224; kk += 32) {
      V16U a, b;
      const __bf16* ap = &sA[n][kk + hi * 8];
      a.q[0] = *(const vi4*)(ap);
      a.q[1] = *(const vi4*)(ap + 16);
      const __bf16* bp = wrow2 + kk + hi * 8;
      b.q[0] = *(const vi4*)(bp);
      b.q[1] = *(const vi4*)(bp + 16);
      accK = __builtin_amdgcn_wmma_f32_16x16x32_bf16(false, a.v, false, b.v,
                                                     (short)0, accK, false, false);
    }
  }

  // ---- Epilogue: bias, soft-threshold (kan only), output scale ----
  float bo  = bias[obase + n];
  float st  = soft_threshold[0];
  float thr = (st > 20.0f) ? st : log1pf(__expf(st));   // softplus
  float osc = output_scale[0];
#pragma unroll
  for (int r = 0; r < 8; ++r) {
    int m = r + hi * 8;                                  // D-matrix layout
    float b  = accB[r] + bo;
    float kv = accK[r];
    float kt = copysignf(fmaxf(fabsf(kv) - thr, 0.0f), kv);
    out[(size_t)(mbase + m) * OUT_F + obase + n] = (b + kt) * osc;
  }
}

// ---------------------------------------------------------------------------
extern "C" void kernel_launch(void* const* d_in, const int* in_sizes, int n_in,
                              void* d_out, int out_size, void* d_ws, size_t ws_size,
                              hipStream_t stream) {
  const float* x   = (const float*)d_in[0];
  const float* bw  = (const float*)d_in[1];
  const float* bb  = (const float*)d_in[2];
  const float* u   = (const float*)d_in[3];
  const float* trn = (const float*)d_in[4];
  const float* scl = (const float*)d_in[5];
  const float* ww  = (const float*)d_in[6];
  const float* sth = (const float*)d_in[7];
  const float* osc = (const float*)d_in[8];
  float* out = (float*)d_out;

  float*  inv_sigma = (float*)d_ws;
  __bf16* Wp        = (__bf16*)((char*)d_ws + 256);   // 16 MB packed weights

  k_sigma<<<1, 1024, 0, stream>>>(bw, u, inv_sigma);

  long total = (long)OUT_F * KTOT;
  k_pack<<<(int)((total + 255) / 256), 256, 0, stream>>>(bw, ww, inv_sigma, Wp);

  dim3 grid(NROWS / 16, OUT_F / 128);   // 1024 x 8 blocks
  k_main<<<grid, 256, 0, stream>>>(x, Wp, bb, trn, scl, sth, osc, out);
}
